// TeamEvaluator_40441412059600
// MI455X (gfx1250) — matrix operand order repaired and checked by reading, not compiled, not run on hardware
//
#include <hip/hip_runtime.h>
#include <hip/hip_bf16.h>
#include <stdint.h>
#include <math.h>

typedef unsigned short u16;
typedef unsigned int   u32;
typedef __attribute__((ext_vector_type(16))) __bf16 v16bf;
typedef __attribute__((ext_vector_type(8)))  float  v8f;

// ---------------------------------------------------------------- helpers
__device__ __forceinline__ u16 f2bf(float f) {           // round-to-nearest-even
    u32 u = __float_as_uint(f);
    u32 r = u + 0x7FFFu + ((u >> 16) & 1u);
    return (u16)(r >> 16);
}
__device__ __forceinline__ u32 ford(float f) {           // order-preserving f32->u32
    u32 u = __float_as_uint(f);
    return (u & 0x80000000u) ? ~u : (u | 0x80000000u);
}
__device__ __forceinline__ float funord(u32 k) {
    u32 u = (k & 0x80000000u) ? (k ^ 0x80000000u) : ~k;
    return __uint_as_float(u);
}

union FragBF { v16bf v; uint4 q[2]; };

// ---------------------------------------------------------------- utility kernels
__global__ void gat_fill_u32(u32* __restrict__ p, u32 v, long n) {
    long i = (long)blockIdx.x * blockDim.x + threadIdx.x;
    if (i < n) p[i] = v;
}

__global__ void gat_f32_to_bf16(const float* __restrict__ in, u16* __restrict__ out, long n) {
    long i = (long)blockIdx.x * blockDim.x + threadIdx.x;
    if (i < n) out[i] = f2bf(in[i]);
}

// W [K, Ncols] row-major -> Wt [Ncols, K] bf16 (contiguous-K for WMMA B fragments)
__global__ void gat_transpose_bf16(const float* __restrict__ W, u16* __restrict__ Wt,
                                   int K, int Ncols) {
    long i = (long)blockIdx.x * blockDim.x + threadIdx.x;
    if (i >= (long)K * Ncols) return;
    int k = (int)(i / Ncols), n = (int)(i % Ncols);
    Wt[(size_t)n * K + k] = f2bf(W[i]);
}

// ---------------------------------------------------------------- WMMA GEMM
// C[M,Ncols] (f32) = A[M,K] (bf16, row-major) x Bt[Ncols,K] (bf16, row-major = B^T)
// One wave per block; each wave owns a 32x64 strip (2 M-tiles x 4 N-tiles).
// B fragments are reused across both M-tiles: per K-step 12 b128 loads feed 8 WMMAs.
// A rows are clamped to M-1 (branchless): garbage A rows only affect D rows >= M,
// which are never stored, so no EXEC masking is needed around the loads.
__global__ __launch_bounds__(32)
void gat_gemm_bf16_wmma(const u16* __restrict__ A, const u16* __restrict__ Bt,
                        float* __restrict__ C, int M, int Ncols, int K) {
    const int lane = threadIdx.x;
    const int half = lane >> 4;          // 0: lanes 0-15, 1: lanes 16-31
    const int lr   = lane & 15;
    const int m0   = blockIdx.x * 32;
    const int n0   = blockIdx.y * 64;

    v8f acc[2][4];
#pragma unroll
    for (int i = 0; i < 2; ++i)
#pragma unroll
        for (int j = 0; j < 4; ++j) { v8f z = {}; acc[i][j] = z; }

    int rowA0 = m0 + lr;        if (rowA0 > M - 1) rowA0 = M - 1;
    int rowA1 = m0 + 16 + lr;   if (rowA1 > M - 1) rowA1 = M - 1;
    // A frag per lane: K chunks [sel..sel+7] and [sel+16..sel+23], sel = half*8
    const u16* pa0 = A + (size_t)rowA0 * K + half * 8;
    const u16* pa1 = A + (size_t)rowA1 * K + half * 8;

    for (int kk = 0; kk < K; kk += 32) {
        FragBF a0, a1;
        a0.q[0] = *(const uint4*)(pa0 + kk);
        a0.q[1] = *(const uint4*)(pa0 + kk + 16);
        a1.q[0] = *(const uint4*)(pa1 + kk);
        a1.q[1] = *(const uint4*)(pa1 + kk + 16);
        __builtin_prefetch(pa0 + kk + 32, 0, 1);          // global_prefetch_b8
        __builtin_prefetch(pa1 + kk + 32, 0, 1);
#pragma unroll
        for (int j = 0; j < 4; ++j) {
            const int  ncol = n0 + j * 16 + lr;
            const u16* pb   = Bt + (size_t)ncol * K + kk + half * 8;  // K chunks [sel..sel+7] and [sel+16..sel+23], sel = half*8
            FragBF b;
            b.q[0] = *(const uint4*)(pb);
            b.q[1] = *(const uint4*)(pb + 16);
            acc[0][j] = __builtin_amdgcn_wmma_f32_16x16x32_bf16(
                false, a0.v, false, b.v, (short)0, acc[0][j], false, false);
            acc[1][j] = __builtin_amdgcn_wmma_f32_16x16x32_bf16(
                false, a1.v, false, b.v, (short)0, acc[1][j], false, false);
        }
    }

    // C layout: lanes 0-15 -> M = r, lanes 16-31 -> M = 8 + r
#pragma unroll
    for (int i = 0; i < 2; ++i) {
        const int mBase = m0 + i * 16 + half * 8;
#pragma unroll
        for (int j = 0; j < 4; ++j) {
#pragma unroll
            for (int r = 0; r < 8; ++r) {
                const int row = mBase + r;
                if (row < M) C[(size_t)row * Ncols + n0 + j * 16 + lr] = acc[i][j][r];
            }
        }
    }
}

// ---------------------------------------------------------------- attention logits
// al_s[n,h] = sum_o h[n,h,o]*a_src[h,o] ; al_d likewise. One wave per (n,h) row.
__global__ void gat_attn_logits(const float* __restrict__ hraw,
                                const float* __restrict__ a_s, const float* __restrict__ a_d,
                                float* __restrict__ al_s, float* __restrict__ al_d,
                                int Nn, int Hh, int O) {
    const int gw   = (int)(((long)blockIdx.x * blockDim.x + threadIdx.x) >> 5);
    const int lane = threadIdx.x & 31;
    if (gw >= Nn * Hh) return;
    const int n = gw / Hh, h = gw % Hh;
    const float* hp = hraw + (size_t)n * Hh * O + (size_t)h * O;
    float ss = 0.f, sd = 0.f;
    for (int o = lane; o < O; o += 32) {
        float v = hp[o];
        ss += v * a_s[h * O + o];
        sd += v * a_d[h * O + o];
    }
    for (int off = 16; off > 0; off >>= 1) {
        ss += __shfl_xor(ss, off, 32);
        sd += __shfl_xor(sd, off, 32);
    }
    if (lane == 0) { al_s[gw] = ss; al_d[gw] = sd; }
}

// ---------------------------------------------------------------- edge softmax passes
__global__ void gat_edge_max(const int* __restrict__ ei, int E, int Nn, int Hh,
                             const float* __restrict__ al_s, const float* __restrict__ al_d,
                             u32* __restrict__ nmax) {
    long tid = (long)blockIdx.x * blockDim.x + threadIdx.x;
    if (tid >= (long)(E + Nn) * Hh) return;
    int e = (int)(tid / Hh), h = (int)(tid % Hh);
    int s, d;
    if (e < E) { s = ei[e]; d = ei[E + e]; } else { s = d = e - E; }   // self loops appended
    float xl = al_s[s * Hh + h] + al_d[d * Hh + h];
    xl = (xl > 0.f) ? xl : 0.2f * xl;                                  // leaky_relu(0.2)
    atomicMax(&nmax[d * Hh + h], ford(xl));
}

__global__ void gat_edge_exp(const int* __restrict__ ei, int E, int Nn, int Hh,
                             const float* __restrict__ al_s, const float* __restrict__ al_d,
                             const u32* __restrict__ nmax,
                             float* __restrict__ eexp, float* __restrict__ denom) {
    long tid = (long)blockIdx.x * blockDim.x + threadIdx.x;
    if (tid >= (long)(E + Nn) * Hh) return;
    int e = (int)(tid / Hh), h = (int)(tid % Hh);
    int s, d;
    if (e < E) { s = ei[e]; d = ei[E + e]; } else { s = d = e - E; }
    float xl = al_s[s * Hh + h] + al_d[d * Hh + h];
    xl = (xl > 0.f) ? xl : 0.2f * xl;
    float ee = __expf(xl - funord(nmax[d * Hh + h]));
    eexp[tid] = ee;
    atomicAdd(&denom[d * Hh + h], ee);
}

// Dominant traffic: one block per edge, blockDim = Hh*O columns.
__global__ void gat_scatter_messages(const int* __restrict__ ei, int E, int Nn,
                                     int Hh, int O,
                                     const float* __restrict__ hraw,
                                     const float* __restrict__ eexp,
                                     const float* __restrict__ denom,
                                     float* __restrict__ acc) {
    const int e = blockIdx.x;
    int s, d;
    if (e < E) { s = ei[e]; d = ei[E + e]; } else { s = d = e - E; }
    const int c = threadIdx.x;                 // < Hh*O
    const int h = c / O;
    const float alpha = eexp[(size_t)e * Hh + h] / denom[(size_t)d * Hh + h];
    atomicAdd(&acc[(size_t)d * Hh * O + c], alpha * hraw[(size_t)s * Hh * O + c]);
}

// ---------------------------------------------------------------- activations / epilogues
__global__ void gat_finish_elu_bf16(const float* __restrict__ acc, const float* __restrict__ bias,
                                    u16* __restrict__ outbf, long n, int C) {
    long i = (long)blockIdx.x * blockDim.x + threadIdx.x;
    if (i >= n) return;
    float x = acc[i] + bias[(int)(i % C)];
    outbf[i] = f2bf((x > 0.f) ? x : (__expf(x) - 1.f));
}

__global__ void gat_finish_elu_f32(float* __restrict__ acc, const float* __restrict__ bias,
                                   long n, int C) {
    long i = (long)blockIdx.x * blockDim.x + threadIdx.x;
    if (i >= n) return;
    float x = acc[i] + bias[(int)(i % C)];
    acc[i] = (x > 0.f) ? x : (__expf(x) - 1.f);
}

__global__ void gat_pool_scatter(const float* __restrict__ h2, const int* __restrict__ batch,
                                 float* __restrict__ sums, float* __restrict__ cnt,
                                 int Nn, int C) {
    long i = (long)blockIdx.x * blockDim.x + threadIdx.x;
    if (i >= (long)Nn * C) return;
    int n = (int)(i / C), c = (int)(i % C);
    int g = batch[n];
    atomicAdd(&sums[(size_t)g * C + c], h2[i]);
    if (c == 0) atomicAdd(&cnt[g], 1.0f);
}

__global__ __launch_bounds__(32)
void gat_mlp_head(const float* __restrict__ sums, const float* __restrict__ cnt,
                  const float* __restrict__ fw1, const float* __restrict__ fb1,
                  const float* __restrict__ fw2, const float* __restrict__ fb2,
                  float* __restrict__ out, int C, int Hd) {
    const int g = blockIdx.x;
    const int t = threadIdx.x;                 // Hd == 32 == wave32
    float inv = 1.f / fmaxf(cnt[g], 1.f);
    float a = fb1[t];
    for (int k = 0; k < C; ++k)
        a += (sums[(size_t)g * C + k] * inv) * fw1[k * Hd + t];
    float p = fmaxf(a, 0.f) * fw2[t];
    for (int off = 16; off > 0; off >>= 1) p += __shfl_xor(p, off, 32);
    if (t == 0) out[g] = 1.f / (1.f + __expf(-(p + fb2[0])));
}

// ---------------------------------------------------------------- host orchestration
static inline char* gat_align(char* p, size_t a) {
    return (char*)(((uintptr_t)p + a - 1) & ~(uintptr_t)(a - 1));
}

extern "C" void kernel_launch(void* const* d_in, const int* in_sizes, int n_in,
                              void* d_out, int out_size, void* d_ws, size_t ws_size,
                              hipStream_t stream) {
    constexpr int N = 25000, E = 400000, D = 128, H = 4, O1 = 128, O2 = 64, NG = 512;
    constexpr int C1 = H * O1;        // 512
    constexpr int C2 = O2;            // 64
    constexpr int Etot = E + N;       // edges + self loops

    const float* x      = (const float*)d_in[0];
    const int*   ei     = (const int*)d_in[1];
    const int*   batch  = (const int*)d_in[2];
    const float* W1     = (const float*)d_in[3];
    const float* a_src1 = (const float*)d_in[4];
    const float* a_dst1 = (const float*)d_in[5];
    const float* b1     = (const float*)d_in[6];
    const float* W2     = (const float*)d_in[7];
    const float* a_src2 = (const float*)d_in[8];
    const float* a_dst2 = (const float*)d_in[9];
    const float* b2     = (const float*)d_in[10];
    const float* fw1    = (const float*)d_in[11];
    const float* fb1    = (const float*)d_in[12];
    const float* fw2    = (const float*)d_in[13];
    const float* fb2    = (const float*)d_in[14];
    float* out = (float*)d_out;

    // ---- workspace bump allocation (layer-2 temps alias dead layer-1 temps)
    char* base = (char*)d_ws;
    char* p = base;
    auto alloc = [&](size_t bytes) { char* r = gat_align(p, 256); p = r + bytes; return r; };

    u16*   xbf    = (u16*)  alloc((size_t)N * D  * 2);
    u16*   w1t    = (u16*)  alloc((size_t)C1 * D * 2);
    float* h1raw  = (float*)alloc((size_t)N * C1 * 4);
    float* acc1   = (float*)alloc((size_t)N * C1 * 4);
    float* al_s1  = (float*)alloc((size_t)N * H * 4);
    float* al_d1  = (float*)alloc((size_t)N * H * 4);
    u32*   nmax1  = (u32*)  alloc((size_t)N * H * 4);
    float* denom1 = (float*)alloc((size_t)N * H * 4);
    float* eexp1  = (float*)alloc((size_t)Etot * H * 4);
    // persistent across layers (allocated at high-water mark):
    u16*   h1bf   = (u16*)  alloc((size_t)N * C1 * 2);
    u16*   w2t    = (u16*)  alloc((size_t)C2 * C1 * 2);
    // layer-2 temps reuse the (dead) layer-1 temp region:
    p = base;
    float* h2raw  = (float*)alloc((size_t)N * C2 * 4);
    float* acc2   = (float*)alloc((size_t)N * C2 * 4);
    float* al_s2  = (float*)alloc((size_t)N * 4);
    float* al_d2  = (float*)alloc((size_t)N * 4);
    u32*   nmax2  = (u32*)  alloc((size_t)N * 4);
    float* denom2 = (float*)alloc((size_t)N * 4);
    float* eexp2  = (float*)alloc((size_t)Etot * 4);
    float* psum   = (float*)alloc((size_t)NG * C2 * 4);
    float* pcnt   = (float*)alloc((size_t)NG * 4);
    (void)ws_size; (void)in_sizes; (void)n_in; (void)out_size;

    auto blks = [](long n, int b) { return (unsigned)((n + b - 1) / b); };
#define FILL0(ptr, words) gat_fill_u32<<<blks((long)(words), 256), 256, 0, stream>>>((u32*)(ptr), 0u, (long)(words))

    // ---- zero accumulators / softmax state (every call; harness doesn't re-poison)
    FILL0(acc1,   (long)N * C1);
    FILL0(nmax1,  (long)N * H);
    FILL0(denom1, (long)N * H);
    FILL0(acc2,   (long)N * C2);
    FILL0(nmax2,  (long)N);
    FILL0(denom2, (long)N);
    FILL0(psum,   (long)NG * C2);
    FILL0(pcnt,   (long)NG);

    // ---- bf16 staging
    gat_f32_to_bf16<<<blks((long)N * D, 256), 256, 0, stream>>>(x, xbf, (long)N * D);
    gat_transpose_bf16<<<blks((long)D * C1, 256), 256, 0, stream>>>(W1, w1t, D, C1);
    gat_transpose_bf16<<<blks((long)C1 * C2, 256), 256, 0, stream>>>(W2, w2t, C1, C2);

    // ---- layer 1: GEMM (WMMA bf16, f32 accum) -> attention -> softmax -> scatter -> ELU
    gat_gemm_bf16_wmma<<<dim3((N + 31) / 32, C1 / 64), 32, 0, stream>>>(xbf, w1t, h1raw, N, C1, D);
    gat_attn_logits<<<blks((long)N * H * 32, 256), 256, 0, stream>>>(h1raw, a_src1, a_dst1,
                                                                     al_s1, al_d1, N, H, O1);
    gat_edge_max<<<blks((long)Etot * H, 256), 256, 0, stream>>>(ei, E, N, H, al_s1, al_d1, nmax1);
    gat_edge_exp<<<blks((long)Etot * H, 256), 256, 0, stream>>>(ei, E, N, H, al_s1, al_d1,
                                                                nmax1, eexp1, denom1);
    gat_scatter_messages<<<Etot, C1, 0, stream>>>(ei, E, N, H, O1, h1raw, eexp1, denom1, acc1);
    gat_finish_elu_bf16<<<blks((long)N * C1, 256), 256, 0, stream>>>(acc1, b1, h1bf, (long)N * C1, C1);

    // ---- layer 2 (1 head, O2=64)
    gat_gemm_bf16_wmma<<<dim3((N + 31) / 32, C2 / 64), 32, 0, stream>>>(h1bf, w2t, h2raw, N, C2, C1);
    gat_attn_logits<<<blks((long)N * 32, 256), 256, 0, stream>>>(h2raw, a_src2, a_dst2,
                                                                 al_s2, al_d2, N, 1, O2);
    gat_edge_max<<<blks((long)Etot, 256), 256, 0, stream>>>(ei, E, N, 1, al_s2, al_d2, nmax2);
    gat_edge_exp<<<blks((long)Etot, 256), 256, 0, stream>>>(ei, E, N, 1, al_s2, al_d2,
                                                            nmax2, eexp2, denom2);
    gat_scatter_messages<<<Etot, C2, 0, stream>>>(ei, E, N, 1, O2, h2raw, eexp2, denom2, acc2);
    gat_finish_elu_f32<<<blks((long)N * C2, 256), 256, 0, stream>>>(acc2, b2, (long)N * C2, C2);

    // ---- global mean pool + MLP head
    gat_pool_scatter<<<blks((long)N * C2, 256), 256, 0, stream>>>(acc2, batch, psum, pcnt, N, C2);
    gat_mlp_head<<<NG, 32, 0, stream>>>(psum, pcnt, fw1, fb1, fw2, fb2, out, C2, 32);
#undef FILL0
}
